// GraphConvolution_30640296690052
// MI455X (gfx1250) — matrix operand order repaired and checked by reading, not compiled, not run on hardware
//
#include <hip/hip_runtime.h>

#define NN   12288
#define INF  256
#define OUTF 128

typedef __attribute__((ext_vector_type(16))) __bf16 v16bf;
typedef __attribute__((ext_vector_type(8)))  __bf16 v8bf;
typedef __attribute__((ext_vector_type(8)))  float  v8f;

// Convert 8 contiguous fp32 values to bf16, optionally adding 1.0 at the
// diagonal position (for adj + I). diag_idx < 0 or >= 8 means "no diagonal".
__device__ __forceinline__ void cvt_chunk(const float* __restrict__ p, int diag_idx,
                                          __bf16* __restrict__ dst) {
  float4 f0 = *(const float4*)(p);
  float4 f1 = *(const float4*)(p + 4);
  float f[8] = {f0.x, f0.y, f0.z, f0.w, f1.x, f1.y, f1.z, f1.w};
  if ((unsigned)diag_idx < 8u) f[diag_idx] += 1.0f;
#pragma unroll
  for (int e = 0; e < 8; ++e) dst[e] = (__bf16)f[e];
}

// ---------------------------------------------------------------------------
// Kernel 1: deg[i] = 1 + sum_j adj[i][j];  dinv[i] = rsqrt(deg[i])
// One 256-thread block (8 wave32) per row.
// ---------------------------------------------------------------------------
__global__ __launch_bounds__(256) void k_deg(const float* __restrict__ adj,
                                             float* __restrict__ dinv) {
  const int row = blockIdx.x;
  const float4* r4 = (const float4*)(adj + (size_t)row * NN);
  float s = 0.0f;
  for (int t = threadIdx.x; t < NN / 4; t += 256) {
    float4 v = r4[t];
    s += (v.x + v.y) + (v.z + v.w);
  }
#pragma unroll
  for (int off = 16; off > 0; off >>= 1) s += __shfl_down(s, off, 32);
  __shared__ float red[8];
  if ((threadIdx.x & 31) == 0) red[threadIdx.x >> 5] = s;
  __syncthreads();
  if (threadIdx.x == 0) {
    float tot = 1.0f;  // self loop
#pragma unroll
    for (int w = 0; w < 8; ++w) tot += red[w];
    dinv[row] = rsqrtf(tot);
  }
}

// ---------------------------------------------------------------------------
// Kernel 2: s2t[o][j] = dinv[j] * (x @ W)[j][o]   (bf16, stored TRANSPOSED)
// W is staged transposed in LDS (64 KB of the 320 KB WGP pool).
// Each of 8 waves owns a 16-row tile of x and all 128 output columns.
// ---------------------------------------------------------------------------
__global__ __launch_bounds__(256) void k_support(const float* __restrict__ x,
                                                 const float* __restrict__ W,
                                                 const float* __restrict__ dinv,
                                                 __bf16* __restrict__ s2t) {
  __shared__ __bf16 wt[OUTF * INF];  // wt[o*INF + k] = W[k][o]
  for (int idx = threadIdx.x; idx < INF * OUTF; idx += 256) {
    int k = idx >> 7;   // 0..255
    int o = idx & 127;  // 0..127
    wt[o * INF + k] = (__bf16)W[idx];
  }
  __syncthreads();

  const int lane = threadIdx.x & 31;
  const int wave = threadIdx.x >> 5;
  const int g    = lane >> 4;
  const int l16  = lane & 15;
  const int jw   = blockIdx.x * 128 + wave * 16;  // row-tile base
  const float* xrow = x + (size_t)(jw + l16) * INF;

  v8f acc[8] = {};
  for (int k0 = 0; k0 < INF; k0 += 32) {
    // A fragment (ISA 16-bit A 16x32 table): lane group g holds
    // K = g*8 + {0..7}  and  K = 16 + g*8 + {0..7}.
    v16bf a;
    cvt_chunk(xrow + k0 + g * 8,      -1, (__bf16*)&a);
    cvt_chunk(xrow + k0 + 16 + g * 8, -1, ((__bf16*)&a) + 8);
#pragma unroll
    for (int t = 0; t < 8; ++t) {
      const int o = t * 16 + l16;
      // B fragment (same order as A): lane group g holds
      // K = g*8 + {0..7}  and  K = 16 + g*8 + {0..7}.
      const int kb = k0 + g * 8;
      v16bf bf;
      *(v8bf*)&bf         = *(const v8bf*)&wt[o * INF + kb];
      *(((v8bf*)&bf) + 1) = *(const v8bf*)&wt[o * INF + kb + 16];
      acc[t] = __builtin_amdgcn_wmma_f32_16x16x32_bf16(false, a, false, bf,
                                                       (short)0, acc[t], false, false);
    }
  }

  // D layout: VGPR v -> row v (lanes 0-15) / v+8 (lanes 16-31), col = l16.
  const int jrow = jw + (g ? 8 : 0);
  float4 d0 = *(const float4*)(dinv + jrow);
  float4 d1 = *(const float4*)(dinv + jrow + 4);
  float dv[8] = {d0.x, d0.y, d0.z, d0.w, d1.x, d1.y, d1.z, d1.w};
#pragma unroll
  for (int t = 0; t < 8; ++t) {
    const int o = t * 16 + l16;
    v8bf sv;
#pragma unroll
    for (int v = 0; v < 8; ++v) sv[v] = (__bf16)(acc[t][v] * dv[v]);
    *(v8bf*)(s2t + (size_t)o * NN + jrow) = sv;  // contiguous in j: one b128 store
  }
}

// ---------------------------------------------------------------------------
// Kernel 3: out[i][o] = dinv[i] * sum_j (adj[i][j]+I) * s2t[o][j] + b[o]
// 384 WGs x 8 waves. Wave owns 1 M-tile (16 rows) x 2 N-tiles (32 cols).
// WG covers 32 rows x 128 cols; adj streamed from HBM exactly once.
// ---------------------------------------------------------------------------
__global__ __launch_bounds__(256) void k_aggregate(const float* __restrict__ adj,
                                                   const __bf16* __restrict__ s2t,
                                                   const float* __restrict__ dinv,
                                                   const float* __restrict__ b,
                                                   float* __restrict__ out) {
  const int lane  = threadIdx.x & 31;
  const int wave  = threadIdx.x >> 5;
  const int g     = lane >> 4;
  const int l16   = lane & 15;
  const int iW    = blockIdx.x * 32 + (wave & 1) * 16;  // M-tile base
  const int tBase = (wave >> 1) * 2;                    // first of 2 N-tiles
  const int i     = iW + l16;                           // this lane's adj row
  const float*  arow  = adj + (size_t)i * NN;
  const __bf16* bcol0 = s2t + (size_t)(tBase * 16 + l16) * NN;
  const __bf16* bcol1 = bcol0 + (size_t)16 * NN;

  v8f acc0 = {}, acc1 = {};
  for (int k0 = 0; k0 < NN; k0 += 32) {
    // A fragment: K-halves interleaved in 8-element granules per lane group.
    const int ka0 = k0 + g * 8;
    const int ka1 = k0 + 16 + g * 8;
    v16bf a;
    cvt_chunk(arow + ka0, i - ka0, (__bf16*)&a);       // adj + I, on the fly
    cvt_chunk(arow + ka1, i - ka1, ((__bf16*)&a) + 8);
    // B fragment: same interleaved 8-element granules as A.
    const int kb = k0 + g * 8;
    v16bf b0, b1;
    *(v8bf*)&b0         = *(const v8bf*)(bcol0 + kb);
    *(((v8bf*)&b0) + 1) = *(const v8bf*)(bcol0 + kb + 16);
    *(v8bf*)&b1         = *(const v8bf*)(bcol1 + kb);
    *(((v8bf*)&b1) + 1) = *(const v8bf*)(bcol1 + kb + 16);
    acc0 = __builtin_amdgcn_wmma_f32_16x16x32_bf16(false, a, false, b0,
                                                   (short)0, acc0, false, false);
    acc1 = __builtin_amdgcn_wmma_f32_16x16x32_bf16(false, a, false, b1,
                                                   (short)0, acc1, false, false);
    // run ahead on the HBM adjacency stream (global_prefetch_b8)
    __builtin_prefetch(arow + k0 + 512, 0, 0);
  }

  const int irow = iW + (g ? 8 : 0);
  float4 d0 = *(const float4*)(dinv + irow);
  float4 d1 = *(const float4*)(dinv + irow + 4);
  float dv[8] = {d0.x, d0.y, d0.z, d0.w, d1.x, d1.y, d1.z, d1.w};
  const int o0 = tBase * 16 + l16;
  const int o1 = o0 + 16;
  const float bb0 = b[o0];
  const float bb1 = b[o1];
#pragma unroll
  for (int v = 0; v < 8; ++v) {
    out[(size_t)(irow + v) * OUTF + o0] = dv[v] * acc0[v] + bb0;
    out[(size_t)(irow + v) * OUTF + o1] = dv[v] * acc1[v] + bb1;
  }
}

// ---------------------------------------------------------------------------
extern "C" void kernel_launch(void* const* d_in, const int* in_sizes, int n_in,
                              void* d_out, int out_size, void* d_ws, size_t ws_size,
                              hipStream_t stream) {
  const float* x   = (const float*)d_in[0];
  const float* adj = (const float*)d_in[1];
  const float* W   = (const float*)d_in[2];
  const float* b   = (const float*)d_in[3];
  float* out = (float*)d_out;

  // workspace layout: [0, 48KB) dinv ; [64KB, 64KB+3MB) s2t (bf16, transposed)
  float*  dinv = (float*)d_ws;
  __bf16* s2t  = (__bf16*)((char*)d_ws + 64 * 1024);

  k_deg      <<<NN,       256, 0, stream>>>(adj, dinv);
  k_support  <<<NN / 128, 256, 0, stream>>>(x, W, dinv, s2t);
  k_aggregate<<<NN / 32,  256, 0, stream>>>(adj, s2t, dinv, b, out);
}
